// GroupAttention_644245094962
// MI455X (gfx1250) — matrix operand order repaired and checked by reading, not compiled, not run on hardware
//
#include <hip/hip_runtime.h>
#include <hip/hip_bf16.h>

typedef __attribute__((ext_vector_type(16))) _Float16 v16h;
typedef __attribute__((ext_vector_type(8)))  float    v8f;

#define B_TOT 256
#define N_TOK 80
#define C_CH  256
#define H_HD  8
#define HD    10
#define M3N   240
#define KPAD  96     // 80 -> 96 (3 WMMA k-steps of 32)
#define HDPAD 32     // 10 -> 32 (1 WMMA k-step)
#define RPE_H 512
#define TROWS 765
#define EPS   5e-5f

// f16 A-fragment K index for element p (=2j+e) of v16h, given lane-half.
__device__ __forceinline__ int a_kidx(int p, int half) {
    return 16 * half + p;
}

// ---------------- CPB MLP -> 16*sigmoid(bias_table) [765*8] ----------------
__global__ void cpb_kernel(const float* __restrict__ table,
                           const float* __restrict__ w1,
                           const float* __restrict__ b1,
                           const float* __restrict__ w2,
                           float* __restrict__ rpb_sig) {
    int tid = blockIdx.x * blockDim.x + threadIdx.x;
    if (tid >= TROWS * H_HD) return;
    int r = tid >> 3, h = tid & 7;
    float t0 = table[r * 2], t1 = table[r * 2 + 1];
    float acc = 0.f;
    for (int j = 0; j < RPE_H; ++j) {
        float hm = fmaxf(0.f, w1[j * 2] * t0 + w1[j * 2 + 1] * t1 + b1[j]);
        acc += hm * w2[h * RPE_H + j];
    }
    rpb_sig[r * H_HD + h] = 16.f / (1.f + __expf(-acc));
}

// ---------------- QKV projection: Y[b,c,m] = sum_n x[b,n,c] w[m,n] + bias --
__global__ __launch_bounds__(256) void qkv_kernel(const float* __restrict__ x,
        const float* __restrict__ w_qkv, const float* __restrict__ q_bias,
        const float* __restrict__ v_bias, float* __restrict__ Y) {
    __shared__ _Float16 x_lds[C_CH * KPAD];   // [c][k]
    __shared__ _Float16 w_lds[M3N * KPAD];    // [m][k]
    int b = blockIdx.x, tid = threadIdx.x;
    for (int i = tid; i < M3N * KPAD; i += 256) {
        int m = i / KPAD, n = i % KPAD;
        w_lds[i] = (n < N_TOK) ? (_Float16)w_qkv[m * N_TOK + n] : (_Float16)0.f;
    }
    const float* xb = x + (size_t)b * N_TOK * C_CH;
    for (int i = tid; i < N_TOK * C_CH; i += 256) {       // coalesced read, LDS transpose
        int n = i / C_CH, c = i % C_CH;
        x_lds[c * KPAD + n] = (_Float16)xb[i];
    }
    for (int i = tid; i < C_CH * (KPAD - N_TOK); i += 256) {
        int c = i / (KPAD - N_TOK), n = N_TOK + i % (KPAD - N_TOK);
        x_lds[c * KPAD + n] = (_Float16)0.f;
    }
    __syncthreads();
    int lane = tid & 31, wave = tid >> 5;
    int half = lane >> 4, l16 = lane & 15;
    float* yb = Y + (size_t)b * C_CH * M3N;
    for (int mb = wave * 2; mb < wave * 2 + 2; ++mb) {
        int row0 = mb * 16;
        v16h af[3];
#pragma unroll
        for (int s = 0; s < 3; ++s)
#pragma unroll
            for (int p = 0; p < 16; ++p)
                af[s][p] = x_lds[(row0 + l16) * KPAD + 32 * s + a_kidx(p, half)];
        for (int nt = 0; nt < 15; ++nt) {
            v8f acc = {};
#pragma unroll
            for (int s = 0; s < 3; ++s) {
                v16h bf;
#pragma unroll
                for (int p = 0; p < 16; ++p)
                    bf[p] = w_lds[(nt * 16 + l16) * KPAD + 32 * s + 16 * half + p];
                acc = __builtin_amdgcn_wmma_f32_16x16x32_f16(false, af[s], false, bf,
                                                             (short)0, acc, false, false);
            }
            int m = nt * 16 + l16;
            float bias = (m < 80) ? q_bias[m] : ((m < 160) ? 0.f : v_bias[m - 160]);
#pragma unroll
            for (int j = 0; j < 8; ++j) {
                int row = row0 + j + 8 * half;
                yb[row * M3N + m] = acc[j] + bias;
            }
        }
    }
}

// ---------------- per-(b,h,c) reciprocal norms of q and k ------------------
__global__ void rnorm_kernel(const float* __restrict__ Y, float* __restrict__ rqk) {
    int idx = blockIdx.x * blockDim.x + threadIdx.x;   // B*H*C threads
    int b = idx >> 11, rem = idx & 2047, h = rem >> 8, c = rem & 255;
    const float* yr = Y + ((size_t)b * C_CH + c) * M3N;
    float sq = 0.f, sk = 0.f;
#pragma unroll
    for (int d = 0; d < HD; ++d) {
        float q = yr[h * HD + d];      sq += q * q;
        float k = yr[80 + h * HD + d]; sk += k * k;
    }
    float* o = rqk + ((size_t)idx) * 2;
    o[0] = 1.f / (sqrtf(sq) + EPS);
    o[1] = 1.f / (sqrtf(sk) + EPS);
}

// ---------------- fused attention per (b,h): logits+bias+softmax+PV --------
__global__ __launch_bounds__(256) void attn_kernel(const float* __restrict__ Y,
        const float* __restrict__ rqk, const float* __restrict__ rpb_sig,
        const int* __restrict__ rel_index, _Float16* __restrict__ aout) {
    __shared__ _Float16 qa[C_CH * HDPAD];     // [c][k]  16 KB
    __shared__ _Float16 ka[C_CH * HDPAD];     // [c][k]  16 KB
    __shared__ _Float16 vt[16 * C_CH];        // [d][c]   8 KB
    __shared__ _Float16 sp[8 * 16 * 272];     // per-wave 16x256 stripe, ~68 KB
    __shared__ float rqs[C_CH], rks[C_CH], ssum[8 * 16];
    int bh = blockIdx.x, b = bh >> 3, h = bh & 7;
    int tid = threadIdx.x;
    const float* yb = Y + (size_t)b * C_CH * M3N;
    for (int i = tid; i < C_CH * HDPAD; i += 256) {
        int c = i >> 5, d = i & 31;
        qa[i] = (d < HD) ? (_Float16)yb[c * M3N + h * HD + d]      : (_Float16)0.f;
        ka[i] = (d < HD) ? (_Float16)yb[c * M3N + 80 + h * HD + d] : (_Float16)0.f;
    }
    for (int i = tid; i < 16 * C_CH; i += 256) {
        int d = i >> 8, c = i & 255;
        vt[i] = (d < HD) ? (_Float16)yb[c * M3N + 160 + h * HD + d] : (_Float16)0.f;
    }
    for (int c = tid; c < C_CH; c += 256) {
        const float* r = rqk + ((size_t)bh * C_CH + c) * 2;
        rqs[c] = r[0]; rks[c] = r[1];
    }
    __syncthreads();
    int lane = tid & 31, wave = tid >> 5;
    int half = lane >> 4, l16 = lane & 15;
    _Float16* spw = sp + wave * 16 * 272;
    for (int sl = 0; sl < 2; ++sl) {
        int rowbase = (wave * 2 + sl) * 16;
        // ---- S = q~ k~^T (K padded 10->32, one WMMA per 16x16 tile) ----
        v16h aq;
#pragma unroll
        for (int p = 0; p < 16; ++p)
            aq[p] = qa[(rowbase + l16) * HDPAD + a_kidx(p, half)];
        for (int nt = 0; nt < 16; ++nt) {
            v16h bf;
#pragma unroll
            for (int p = 0; p < 16; ++p)
                bf[p] = ka[(nt * 16 + l16) * HDPAD + 16 * half + p];
            v8f acc = {};
            acc = __builtin_amdgcn_wmma_f32_16x16x32_f16(false, aq, false, bf,
                                                         (short)0, acc, false, false);
            int col = nt * 16 + l16;
            float rkc = rks[col];
#pragma unroll
            for (int j = 0; j < 8; ++j) {
                int row = rowbase + j + 8 * half;
                float sv = acc[j] * rqs[row] * rkc
                         + rpb_sig[rel_index[row * C_CH + col] * H_HD + h];
                spw[(j + 8 * half) * 272 + col] = (_Float16)sv;
            }
        }
        __syncthreads();
        // ---- softmax over each 256-wide row (2 lanes per row, in place) ----
        {
            int row = lane >> 1, seg = lane & 1;
            _Float16* sr = spw + row * 272 + seg * 128;
            float m = -1e30f;
            for (int i = 0; i < 128; ++i) m = fmaxf(m, (float)sr[i]);
            m = fmaxf(m, __shfl_xor(m, 1, 32));
            float s = 0.f;
            for (int i = 0; i < 128; ++i) {
                float e = __expf((float)sr[i] - m);
                sr[i] = (_Float16)e;            // unnormalized prob (<=1, f16-safe)
                s += e;
            }
            s += __shfl_xor(s, 1, 32);
            if (seg == 0) ssum[wave * 16 + row] = 1.f / s;
        }
        __syncthreads();
        // ---- O = P V  (K = 256 over 8 WMMA k-steps; 1/sum folded here) ----
        v8f o = {};
        for (int s = 0; s < 8; ++s) {
            v16h ap, bv;
#pragma unroll
            for (int p = 0; p < 16; ++p)
                ap[p] = spw[l16 * 272 + 32 * s + a_kidx(p, half)];
#pragma unroll
            for (int p = 0; p < 16; ++p)
                bv[p] = vt[l16 * C_CH + 32 * s + 16 * half + p];
            o = __builtin_amdgcn_wmma_f32_16x16x32_f16(false, ap, false, bv,
                                                       (short)0, o, false, false);
        }
        int d = l16;
        if (d < HD) {
#pragma unroll
            for (int j = 0; j < 8; ++j) {
                int row = rowbase + j + 8 * half;
                float inv = ssum[wave * 16 + j + 8 * half];
                aout[((size_t)b * C_CH + row) * N_TOK + h * HD + d] = (_Float16)(o[j] * inv);
            }
        }
        __syncthreads();
    }
}

// ---------------- output projection + transpose to [B,N,C] ----------------
__global__ __launch_bounds__(256) void proj_kernel(const _Float16* __restrict__ aout,
        const float* __restrict__ proj_w, const float* __restrict__ proj_b,
        float* __restrict__ out) {
    __shared__ _Float16 a_lds[C_CH * KPAD];   // [c][k]
    __shared__ _Float16 pw[N_TOK * KPAD];     // [n][k]
    int b = blockIdx.x, tid = threadIdx.x;
    const _Float16* ab = aout + (size_t)b * C_CH * N_TOK;
    for (int i = tid; i < C_CH * N_TOK; i += 256) {
        int c = i / N_TOK, m = i % N_TOK;
        a_lds[c * KPAD + m] = ab[i];
    }
    for (int i = tid; i < C_CH * (KPAD - N_TOK); i += 256) {
        int c = i / (KPAD - N_TOK), m = N_TOK + i % (KPAD - N_TOK);
        a_lds[c * KPAD + m] = (_Float16)0.f;
    }
    for (int i = tid; i < N_TOK * KPAD; i += 256) {
        int n = i / KPAD, k = i % KPAD;
        pw[i] = (k < N_TOK) ? (_Float16)proj_w[n * N_TOK + k] : (_Float16)0.f;
    }
    __syncthreads();
    int lane = tid & 31, wave = tid >> 5;
    int half = lane >> 4, l16 = lane & 15;
    for (int mb = wave * 2; mb < wave * 2 + 2; ++mb) {
        int row0 = mb * 16;
        v16h af[3];
#pragma unroll
        for (int s = 0; s < 3; ++s)
#pragma unroll
            for (int p = 0; p < 16; ++p)
                af[s][p] = a_lds[(row0 + l16) * KPAD + 32 * s + a_kidx(p, half)];
        for (int nt = 0; nt < 5; ++nt) {
            v8f acc = {};
#pragma unroll
            for (int s = 0; s < 3; ++s) {
                v16h bf;
#pragma unroll
                for (int p = 0; p < 16; ++p)
                    bf[p] = pw[(nt * 16 + l16) * KPAD + 32 * s + 16 * half + p];
                acc = __builtin_amdgcn_wmma_f32_16x16x32_f16(false, af[s], false, bf,
                                                             (short)0, acc, false, false);
            }
            int n = nt * 16 + l16;
            float pb = proj_b[n];
#pragma unroll
            for (int j = 0; j < 8; ++j) {
                int c = row0 + j + 8 * half;
                out[((size_t)b * N_TOK + n) * C_CH + c] = acc[j] + pb;  // [B,N,C]
            }
        }
    }
}

extern "C" void kernel_launch(void* const* d_in, const int* in_sizes, int n_in,
                              void* d_out, int out_size, void* d_ws, size_t ws_size,
                              hipStream_t stream) {
    (void)in_sizes; (void)n_in; (void)out_size; (void)ws_size;
    const float* x      = (const float*)d_in[0];
    const float* w_qkv  = (const float*)d_in[1];
    const float* q_bias = (const float*)d_in[2];
    const float* v_bias = (const float*)d_in[3];
    const float* w1     = (const float*)d_in[4];
    const float* b1     = (const float*)d_in[5];
    const float* w2     = (const float*)d_in[6];
    const float* proj_w = (const float*)d_in[7];
    const float* proj_b = (const float*)d_in[8];
    const float* table  = (const float*)d_in[9];
    const int*   rel    = (const int*)d_in[10];

    char* ws = (char*)d_ws;
    size_t off = 0;
    float* Y = (float*)(ws + off);        off += (size_t)B_TOT * C_CH * M3N * 4;   // 62.9 MB
    float* rqk = (float*)(ws + off);      off += (size_t)B_TOT * H_HD * C_CH * 8;  //  4.2 MB
    float* rpb = (float*)(ws + off);      off += (size_t)TROWS * H_HD * 4;         //   24 KB
    _Float16* aout = (_Float16*)(ws + off); // + 10.5 MB

    cpb_kernel<<<(TROWS * H_HD + 255) / 256, 256, 0, stream>>>(table, w1, b1, w2, rpb);
    qkv_kernel<<<B_TOT, 256, 0, stream>>>(x, w_qkv, q_bias, v_bias, Y);
    rnorm_kernel<<<(B_TOT * H_HD * C_CH) / 256, 256, 0, stream>>>(Y, rqk);
    attn_kernel<<<B_TOT * H_HD, 256, 0, stream>>>(Y, rqk, rpb, rel, aout);
    proj_kernel<<<B_TOT, 256, 0, stream>>>(aout, proj_w, proj_b, (float*)d_out);
}
